// TokenChoiceTopKRouter_46334107189527
// MI455X (gfx1250) — compile-verified, matrix operand order checked by reading, not run on hardware
//
#include <hip/hip_runtime.h>
#include <hip/hip_bf16.h>
#include <cstdint>

// ---------------------------------------------------------------------------
// MoE token-choice top-k router for MI455X (gfx1250, wave32, WMMA).
//   logits = x (T,4096) @ gate_w^T (4096,64)   -> bf16 WMMA, fp32 accum
//   softmax(fp32) -> top-8 -> stable counting sort by expert id
// Outputs (concatenated in d_out): scores_sorted (T*8 f32),
//   token_indices (T*8 i32), num_tokens_per_expert (64 i32)
// ---------------------------------------------------------------------------

typedef __attribute__((ext_vector_type(16))) __bf16 v16bf;
typedef __attribute__((ext_vector_type(8)))  float  v8f;

#define DIM   4096
#define NEXP  64
#define TOPK  8
#define TPB   128      // tokens per block (8 waves x 16 tokens)
#define KSTEP 32       // K per v_wmma_f32_16x16x32_bf16

// Load 8+8 contiguous fp32 (two 32B runs) and convert to the 16-bf16 fragment
// order required by the CDNA5 16-bit A/B VGPR layout (elements 0..7 = run at
// k=half*8, elements 8..15 = run at k=16+half*8).
__device__ __forceinline__ v16bf load_frag16(const float* __restrict__ p,
                                             int kA, int kB) {
  const float4* q0 = (const float4*)(p + kA);
  const float4* q1 = (const float4*)(p + kB);
  float4 f0 = q0[0], f1 = q0[1], f2 = q1[0], f3 = q1[1];
  v16bf r;
  r[0]=(__bf16)f0.x; r[1]=(__bf16)f0.y; r[2]=(__bf16)f0.z; r[3]=(__bf16)f0.w;
  r[4]=(__bf16)f1.x; r[5]=(__bf16)f1.y; r[6]=(__bf16)f1.z; r[7]=(__bf16)f1.w;
  r[8]=(__bf16)f2.x; r[9]=(__bf16)f2.y; r[10]=(__bf16)f2.z; r[11]=(__bf16)f2.w;
  r[12]=(__bf16)f3.x; r[13]=(__bf16)f3.y; r[14]=(__bf16)f3.z; r[15]=(__bf16)f3.w;
  return r;
}

// Pass 1: fused gate GEMM (WMMA) + softmax + top-8 + per-block histogram.
__global__ __launch_bounds__(256) void router_gemm_topk(
    const float* __restrict__ x, const float* __restrict__ gw,
    float* __restrict__ topScore, int* __restrict__ topExp,
    int* __restrict__ blockHist) {
  __shared__ float logits[TPB * 65];   // pad 64->65: conflict-free row reads
  __shared__ int   hist[NEXP];

  const int tid  = threadIdx.x;
  const int wave = tid >> 5;
  const int lane = tid & 31;
  const int m16  = lane & 15;   // row-in-tile
  const int half = lane >> 4;   // lane half selects K sub-runs
  const int mBase = blockIdx.x * TPB + wave * 16;

  if (tid < NEXP) hist[tid] = 0;

  v8f acc0 = {}, acc1 = {}, acc2 = {}, acc3 = {};

  const float* aRow  = x  + (size_t)(mBase + m16) * DIM;
  const float* bRow0 = gw + (size_t)( 0 + m16) * DIM;
  const float* bRow1 = gw + (size_t)(16 + m16) * DIM;
  const float* bRow2 = gw + (size_t)(32 + m16) * DIM;
  const float* bRow3 = gw + (size_t)(48 + m16) * DIM;

  for (int k0 = 0; k0 < DIM; k0 += KSTEP) {
    const int kA = k0 + half * 8;
    const int kB = kA + 16;
    // gfx1250 global_prefetch of next A chunk (x is the HBM-bound stream)
    if (k0 + KSTEP < DIM) __builtin_prefetch(aRow + k0 + KSTEP, 0, 3);

    v16bf a  = load_frag16(aRow,  kA, kB);
    v16bf b0 = load_frag16(bRow0, kA, kB);
    v16bf b1 = load_frag16(bRow1, kA, kB);
    v16bf b2 = load_frag16(bRow2, kA, kB);
    v16bf b3 = load_frag16(bRow3, kA, kB);

    acc0 = __builtin_amdgcn_wmma_f32_16x16x32_bf16(false, a, false, b0,
                                                   (short)0, acc0, false, false);
    acc1 = __builtin_amdgcn_wmma_f32_16x16x32_bf16(false, a, false, b1,
                                                   (short)0, acc1, false, false);
    acc2 = __builtin_amdgcn_wmma_f32_16x16x32_bf16(false, a, false, b2,
                                                   (short)0, acc2, false, false);
    acc3 = __builtin_amdgcn_wmma_f32_16x16x32_bf16(false, a, false, b3,
                                                   (short)0, acc3, false, false);
  }

  // C/D layout: VGPR r holds token m = r + 8*half; expert n = 16*t + m16.
  #pragma unroll
  for (int r = 0; r < 8; ++r) {
    const int row = wave * 16 + r + 8 * half;
    logits[row * 65 +  0 + m16] = acc0[r];
    logits[row * 65 + 16 + m16] = acc1[r];
    logits[row * 65 + 32 + m16] = acc2[r];
    logits[row * 65 + 48 + m16] = acc3[r];
  }
  __syncthreads();

  // One thread per token: softmax (fp32) + 8-pass argmax (stable tie-break).
  if (tid < TPB) {
    const float* row = &logits[tid * 65];
    float mx = -INFINITY;
    #pragma unroll 8
    for (int e = 0; e < NEXP; ++e) mx = fmaxf(mx, row[e]);
    float sum = 0.f;
    #pragma unroll 8
    for (int e = 0; e < NEXP; ++e) sum += __expf(row[e] - mx);
    const float inv = 1.0f / sum;

    unsigned long long taken = 0ull;
    const int gtok = blockIdx.x * TPB + tid;
    for (int j = 0; j < TOPK; ++j) {
      float best = -INFINITY; int bi = 0;
      for (int e = 0; e < NEXP; ++e) {
        float v = row[e];
        bool sel = (v > best) && !((taken >> e) & 1ull);
        best = sel ? v : best;
        bi   = sel ? e : bi;
      }
      taken |= (1ull << bi);
      topScore[(size_t)gtok * TOPK + j] = __expf(best - mx) * inv;
      topExp[(size_t)gtok * TOPK + j]   = bi;
      atomicAdd(&hist[bi], 1);
    }
  }
  __syncthreads();
  if (tid < NEXP) blockHist[blockIdx.x * NEXP + tid] = hist[tid];
}

// Pass 2: per-(expert, block) exclusive prefix + global expert bases + counts.
__global__ __launch_bounds__(NEXP) void router_offsets(
    const int* __restrict__ blockHist, int* __restrict__ blockBase,
    int* __restrict__ outCnt, int nBlocks) {
  __shared__ int tot[NEXP];
  __shared__ int base[NEXP];
  const int e = threadIdx.x;
  int run = 0;
  for (int b = 0; b < nBlocks; ++b) {
    blockBase[e * nBlocks + b] = run;
    run += blockHist[b * NEXP + e];
  }
  tot[e] = run;
  __syncthreads();
  if (e == 0) {
    int acc = 0;
    for (int i = 0; i < NEXP; ++i) { base[i] = acc; acc += tot[i]; }
  }
  __syncthreads();
  outCnt[e] = tot[e];
  const int be = base[e];
  for (int b = 0; b < nBlocks; ++b) blockBase[e * nBlocks + b] += be;
}

// Pass 3: stable in-block scatter (thread == expert scans flat order).
__global__ __launch_bounds__(NEXP) void router_scatter(
    const float* __restrict__ topScore, const int* __restrict__ topExp,
    const int* __restrict__ blockBase, float* __restrict__ outScore,
    int* __restrict__ outTok, int nBlocks) {
  __shared__ float sScore[TPB * TOPK];
  __shared__ int   sExp[TPB * TOPK];
  const int nFlat = TPB * TOPK;                    // 1024
  const int fBase = blockIdx.x * nFlat;
  for (int i = threadIdx.x; i < nFlat; i += blockDim.x) {
    sScore[i] = topScore[fBase + i];
    sExp[i]   = topExp[fBase + i];
  }
  __syncthreads();
  const int e = threadIdx.x;
  int pos = blockBase[e * nBlocks + blockIdx.x];
  for (int i = 0; i < nFlat; ++i) {
    if (sExp[i] == e) {
      outScore[pos] = sScore[i];
      outTok[pos]   = (fBase + i) / TOPK;          // argsort//TOPK == token id
      ++pos;
    }
  }
}

extern "C" void kernel_launch(void* const* d_in, const int* in_sizes, int n_in,
                              void* d_out, int out_size, void* d_ws, size_t ws_size,
                              hipStream_t stream) {
  const float* x  = (const float*)d_in[0];
  const float* gw = (const float*)d_in[1];
  const int T = in_sizes[0] / DIM;           // 16384
  const int nBlocks = T / TPB;               // 128

  // Workspace layout
  float* topScore  = (float*)d_ws;                                   // T*8 f32
  int*   topExp    = (int*)((char*)d_ws + (size_t)T * TOPK * 4);     // T*8 i32
  int*   blockHist = (int*)((char*)d_ws + (size_t)T * TOPK * 8);     // nBlocks*64
  int*   blockBase = blockHist + nBlocks * NEXP;                     // 64*nBlocks

  // Output layout (concatenated, 4-byte elements)
  float* outScore = (float*)d_out;
  int*   outTok   = (int*)d_out + (size_t)T * TOPK;
  int*   outCnt   = (int*)d_out + (size_t)2 * T * TOPK;

  router_gemm_topk<<<nBlocks, 256, 0, stream>>>(x, gw, topScore, topExp,
                                                blockHist);
  router_offsets<<<1, NEXP, 0, stream>>>(blockHist, blockBase, outCnt, nBlocks);
  router_scatter<<<nBlocks, NEXP, 0, stream>>>(topScore, topExp, blockBase,
                                               outScore, outTok, nBlocks);
}
